// AdjacencyMatchingLoss_816043786442
// MI455X (gfx1250) — matrix operand order repaired and checked by reading, not compiled, not run on hardware
//
#include <hip/hip_runtime.h>
#include <hip/hip_bf16.h>
#include <stdint.h>

// Problem sizes (fixed by the reference setup_inputs()).
#define B_DIM   16
#define N_LOG   512
#define N_PHYS  2048
#define E_DIM   2048
#define EPS_F   1e-8f

typedef __attribute__((ext_vector_type(16))) __bf16 v16bf;
typedef __attribute__((ext_vector_type(8)))  float  v8f;

union Frag16 { v16bf v; uint4 u[2]; };

// ---------------- helpers ----------------
__device__ __forceinline__ uint16_t f32_to_bf16_rne(float x) {
  uint32_t u = __float_as_uint(x);
  uint32_t r = u + 0x7FFFu + ((u >> 16) & 1u);
  return (uint16_t)(r >> 16);
}
__device__ __forceinline__ float bf16_bits_to_f32(uint16_t h) {
  return __uint_as_float(((uint32_t)h) << 16);
}

// CDNA5 async global->LDS copy, 16B per lane.
// lds_ptr is a generic pointer to shared memory; on AMDGPU its low 32 bits are
// the wave-relative LDS offset (see ISA 10.2), which is
// exactly what the instruction's LDS-address VGPR wants. Passing the
// pointer-derived value into the asm also makes the __shared__ object
// address-taken, so the "memory" clobber correctly covers the LDS writes.
__device__ __forceinline__ void async_cp16(void* lds_ptr, const void* gptr) {
  unsigned loff = (unsigned)(uintptr_t)lds_ptr;
  asm volatile("global_load_async_to_lds_b128 %0, %1, off"
               :: "v"(loff), "v"(gptr) : "memory");
}

// ---------------- K0: zero accumulators ----------------
__global__ void k_zero(float* acc) {
  if (threadIdx.x < 32) acc[threadIdx.x] = 0.0f;
}

// ---------------- K1: A_t[q][p] = bf16(d_hw[p][q] == 1) ----------------
__global__ __launch_bounds__(1024) void k_build_At(const int* __restrict__ d_hw,
                                                   uint16_t* __restrict__ At) {
  __shared__ uint16_t t[32][33];
  const int tx = threadIdx.x, ty = threadIdx.y;
  const int p0 = blockIdx.y * 32, q0 = blockIdx.x * 32;
  const int d = d_hw[(size_t)(p0 + ty) * N_PHYS + (q0 + tx)];
  t[ty][tx] = (d == 1) ? (uint16_t)0x3F80u : (uint16_t)0u;   // bf16 1.0 / 0.0
  __syncthreads();
  At[(size_t)(q0 + ty) * N_PHYS + (p0 + tx)] = t[tx][ty];
}

// ---------------- K2: split P (f32) into hi/lo bf16 ----------------
__global__ void k_split(const float* __restrict__ P, uint16_t* __restrict__ hi,
                        uint16_t* __restrict__ lo, size_t n) {
  size_t i = (size_t)blockIdx.x * blockDim.x + threadIdx.x;
  if (i >= n) return;
  float x = P[i];
  uint16_t h = f32_to_bf16_rne(x);
  float rem = x - bf16_bits_to_f32(h);      // exact residual
  hi[i] = h;
  lo[i] = f32_to_bf16_rne(rem);
}

// ---------------- K3: PA[b] = P[b] * A_hw  via bf16 WMMA (hi+lo) ----------------
// Block tile 128(M) x 128(N), BK=32, 8 waves (256 thr), wave tile 64x32.
// LDS per buffer: P_hi[128][32] + P_lo[128][32] + At[128 cols][32 k] bf16 = 24KB, x2 buffers.
__global__ __launch_bounds__(256) void k_gemm(const uint16_t* __restrict__ Phi,
                                              const uint16_t* __restrict__ Plo,
                                              const uint16_t* __restrict__ At,
                                              float* __restrict__ PA) {
  __shared__ __align__(16) unsigned char smem[2 * 24576];

  const int tid  = threadIdx.x;
  const int lane = tid & 31;
  const int wave = tid >> 5;
  const int wm = wave >> 2;            // 0..1 -> 64-row half
  const int wn = wave & 3;             // 0..3 -> 32-col quarter
  const int lh  = lane >> 4;           // lane half (K-split select)
  const int l16 = lane & 15;

  const int q0 = blockIdx.x * 128;     // N offset
  const int i0 = blockIdx.y * 128;     // M offset
  const int bz = blockIdx.z;           // batch

  const int KT = N_PHYS / 32;

  auto issue = [&](int bufi, int kt) {
    unsigned char* lb = smem + bufi * 24576;
    const size_t k0 = (size_t)kt * 32;
#pragma unroll
    for (int h = 0; h < 2; ++h) {
      const int s   = tid + h * 256;       // 0..511
      const int row = s >> 2;              // 0..127 (P row / At col)
      const int seg = s & 3;               // 16B segment within 64B line
      unsigned char* lp = lb + row * 64 + seg * 16;
      const size_t gp = ((size_t)(bz * N_LOG + i0 + row)) * N_PHYS + k0 + (size_t)seg * 8;
      const size_t ga = ((size_t)(q0 + row)) * N_PHYS + k0 + (size_t)seg * 8;
      async_cp16(lp,          Phi + gp);
      async_cp16(lp + 8192,   Plo + gp);
      async_cp16(lp + 16384,  At  + ga);
    }
  };

  v8f acc[4][2] = {};

  issue(0, 0);
#pragma unroll 1
  for (int kt = 0; kt < KT; ++kt) {
    if (kt + 1 < KT) {
      issue((kt + 1) & 1, kt + 1);
      asm volatile("s_wait_asynccnt 6" ::: "memory");   // current buffer's 6 done
    } else {
      asm volatile("s_wait_asynccnt 0" ::: "memory");
    }
    __syncthreads();

    const unsigned char* base = smem + (kt & 1) * 24576;
    const unsigned char* phiB = base;
    const unsigned char* ploB = base + 8192;
    const unsigned char* atB  = base + 16384;

    // B fragments (A_hw, KxN=32x16): lane = column n, u[0]=K(0-7|8-15), u[1]=K(16-23|24-31).
    Frag16 fb[2];
#pragma unroll
    for (int ni = 0; ni < 2; ++ni) {
      const unsigned char* p = atB + (wn * 32 + ni * 16 + l16) * 64 + lh * 16;
      fb[ni].u[0] = *(const uint4*)(p);
      fb[ni].u[1] = *(const uint4*)(p + 32);
    }
#pragma unroll
    for (int mi = 0; mi < 4; ++mi) {
      // A fragments (P, MxK=16x32): lane = row m, u[0]=K(0-7|8-15), u[1]=K(16-23|24-31).
      const unsigned ro = (unsigned)((wm * 64 + mi * 16 + l16) * 64 + lh * 16);
      Frag16 fh, fl;
      fh.u[0] = *(const uint4*)(phiB + ro);
      fh.u[1] = *(const uint4*)(phiB + ro + 32);
      fl.u[0] = *(const uint4*)(ploB + ro);
      fl.u[1] = *(const uint4*)(ploB + ro + 32);
#pragma unroll
      for (int ni = 0; ni < 2; ++ni) {
        acc[mi][ni] = __builtin_amdgcn_wmma_f32_16x16x32_bf16(
            false, fh.v, false, fb[ni].v, (short)0, acc[mi][ni], false, false);
        acc[mi][ni] = __builtin_amdgcn_wmma_f32_16x16x32_bf16(
            false, fl.v, false, fb[ni].v, (short)0, acc[mi][ni], false, false);
      }
    }
    __syncthreads();   // all reads of this buffer done before it is overwritten
  }

  // Epilogue: C/D layout -> VGPR r holds M=r (lanes 0-15) / M=r+8 (lanes 16-31), N=lane%16.
  float* PAb = PA + (size_t)bz * N_LOG * N_PHYS;
#pragma unroll
  for (int mi = 0; mi < 4; ++mi) {
#pragma unroll
    for (int ni = 0; ni < 2; ++ni) {
      const int mbase = i0 + wm * 64 + mi * 16 + lh * 8;
      const int ncol  = q0 + wn * 32 + ni * 16 + l16;
#pragma unroll
      for (int r = 0; r < 8; ++r)
        PAb[(size_t)(mbase + r) * N_PHYS + ncol] = acc[mi][ni][r];
    }
  }
}

// ---------------- K4: per-edge weighted dot, reduced per batch ----------------
__global__ __launch_bounds__(256) void k_edge(const float* __restrict__ PA,
                                              const float* __restrict__ P,
                                              const int* __restrict__ esrc,
                                              const int* __restrict__ edst,
                                              const float* __restrict__ ew,
                                              float* __restrict__ acc) {
  const int lane = threadIdx.x & 31;
  const int wave = threadIdx.x >> 5;
  const int b = blockIdx.y;
  const int e = blockIdx.x * 8 + wave;
  const int s = esrc[b * E_DIM + e];
  const int d = edst[b * E_DIM + e];
  const float w = ew[b * E_DIM + e];
  const float4* gi = (const float4*)(PA + ((size_t)b * N_LOG + s) * N_PHYS);
  const float4* gj = (const float4*)(P  + ((size_t)b * N_LOG + d) * N_PHYS);
  float sum = 0.0f;
#pragma unroll 4
  for (int it = 0; it < 16; ++it) {
    const int idx = it * 32 + lane;
    const float4 a = gi[idx], c = gj[idx];
    sum += a.x * c.x + a.y * c.y + a.z * c.z + a.w * c.w;
  }
  for (int off = 16; off > 0; off >>= 1) sum += __shfl_down(sum, off, 32);
  if (lane == 0) {
    atomicAdd(&acc[b], w * sum);       // sample_adj[b]
    atomicAdd(&acc[16 + b], w);        // sum of weights
  }
}

// ---------------- K5: loss = -mean(adj / max(sw, eps)) ----------------
__global__ void k_final(const float* __restrict__ acc, float* __restrict__ out) {
  if (threadIdx.x == 0) {
    float s = 0.0f;
    for (int b = 0; b < B_DIM; ++b) {
      float sw = acc[16 + b];
      s += acc[b] / fmaxf(sw, EPS_F);
    }
    out[0] = -s / (float)B_DIM;
  }
}

// ---------------- host launch ----------------
extern "C" void kernel_launch(void* const* d_in, const int* in_sizes, int n_in,
                              void* d_out, int out_size, void* d_ws, size_t ws_size,
                              hipStream_t stream) {
  (void)in_sizes; (void)n_in; (void)out_size; (void)ws_size;
  const float* P  = (const float*)d_in[0];
  const int* dhw  = (const int*)d_in[1];
  const int* esrc = (const int*)d_in[2];
  const int* edst = (const int*)d_in[3];
  const float* ew = (const float*)d_in[4];
  float* out = (float*)d_out;

  unsigned char* ws = (unsigned char*)d_ws;
  uint16_t* At  = (uint16_t*)(ws);                          //  8 MB  A_hw^T bf16
  uint16_t* Phi = (uint16_t*)(ws + (8ull  << 20));          // 32 MB  P hi
  uint16_t* Plo = (uint16_t*)(ws + (40ull << 20));          // 32 MB  P lo
  float*    PA  = (float*)   (ws + (72ull << 20));          // 64 MB  PA f32
  float*    acc = (float*)   (ws + (136ull << 20));         // 128 B  reductions

  hipLaunchKernelGGL(k_zero, dim3(1), dim3(32), 0, stream, acc);
  hipLaunchKernelGGL(k_build_At, dim3(N_PHYS / 32, N_PHYS / 32), dim3(32, 32),
                     0, stream, dhw, At);
  const size_t nP = (size_t)B_DIM * N_LOG * N_PHYS;
  hipLaunchKernelGGL(k_split, dim3((unsigned)(nP / 256)), dim3(256), 0, stream,
                     P, Phi, Plo, nP);
  hipLaunchKernelGGL(k_gemm, dim3(N_PHYS / 128, N_LOG / 128, B_DIM), dim3(256),
                     0, stream, Phi, Plo, At, PA);
  hipLaunchKernelGGL(k_edge, dim3(E_DIM / 8, B_DIM), dim3(256), 0, stream,
                     PA, P, esrc, edst, ew, acc);
  hipLaunchKernelGGL(k_final, dim3(1), dim3(1), 0, stream, acc, out);
}
